// EdgeNetwork_3255585210370
// MI455X (gfx1250) — matrix operand order repaired and checked by reading, not compiled, not run on hardware
//
#include <hip/hip_runtime.h>

// ---------------- problem constants (from reference) ----------------
#define N_EDGES_C 800000
#define HID_C     128
#define NLAY      3
#define LNEPS     1e-5f

// ---------------- tiling ----------------
#define NWAVE 4      // waves per block
#define MTILE 32     // edges per wave = two 16-row WMMA M-tiles

// ---------------- padded LDS row strides (bank-conflict free: bank = row*4 mod 64) ----
#define WSTRIDE 136  // W tile row stride in bf16 elems (272 B, 16B aligned)
#define FSTRIDE 132  // f32 h-tile row stride in f32 elems (528 B, 16B aligned)
#define BSTRIDE 136  // bf16 h-tile row stride in bf16 elems (272 B, 16B aligned)

typedef __attribute__((ext_vector_type(4)))  __bf16 v4bf;
typedef __attribute__((ext_vector_type(8)))  __bf16 v8bf;
typedef __attribute__((ext_vector_type(16))) __bf16 v16bf;
typedef __attribute__((ext_vector_type(8)))  float  v8f;

// W [L][K][N] f32  ->  Wt [L][N][K] bf16 (transposed, dense; padding added at LDS fill)
__global__ void prep_w_bf16(const float* __restrict__ W, unsigned short* __restrict__ Wt) {
  int idx = blockIdx.x * blockDim.x + threadIdx.x;
  if (idx >= NLAY * HID_C * HID_C) return;
  int l   = idx / (HID_C * HID_C);
  int rem = idx - l * HID_C * HID_C;
  int n   = rem / HID_C;
  int k   = rem - n * HID_C;
  __bf16 b = (__bf16)W[(l * HID_C + k) * HID_C + n];
  Wt[idx] = __builtin_bit_cast(unsigned short, b);
}

__device__ __forceinline__ float wave_sum32(float v) {
#pragma unroll
  for (int m = 16; m >= 1; m >>= 1) v += __shfl_xor(v, m, 32);
  return v;
}

__device__ __forceinline__ v16bf cat16(v8bf lo, v8bf hi) {
  return __builtin_shufflevector(lo, hi, 0,1,2,3,4,5,6,7,8,9,10,11,12,13,14,15);
}

__global__ void __launch_bounds__(NWAVE * 32)
edge_net_kernel(const float* __restrict__ x,
                const long long* __restrict__ eidx,          // [2][E] int64
                const unsigned short* __restrict__ WtRaw,    // [L][N][K] bf16 bits (dense)
                const float* __restrict__ bs,                // [L][HID]
                const float* __restrict__ gammas,            // [L][HID]
                const float* __restrict__ betas,             // [L][HID]
                const float* __restrict__ w_out,             // [HID]
                const float* __restrict__ b_out,             // [1]
                float* __restrict__ out)                     // [E]
{
  extern __shared__ __align__(16) char smem_raw[];
  __bf16* sWt = (__bf16*)smem_raw;                                  // 128 x WSTRIDE bf16
  float*  sHf = (float*)(smem_raw + HID_C * WSTRIDE * 2);           // NWAVE x 32 x FSTRIDE f32
  __bf16* sHb = (__bf16*)(smem_raw + HID_C * WSTRIDE * 2
                          + NWAVE * MTILE * FSTRIDE * 4);           // NWAVE x 32 x BSTRIDE bf16

  const int tid  = threadIdx.x;
  const int lane = tid & 31;
  const int wave = tid >> 5;
  const int hi   = lane >> 4;      // lane half
  const int row  = lane & 15;      // M-row / N-col index within a 16-wide tile

  float*  hf = sHf + wave * MTILE * FSTRIDE;   // GEMM output / LN scratch (f32)
  __bf16* hb = sHb + wave * MTILE * BSTRIDE;   // A-matrix source (bf16)

  const long long tileBase = ((long long)blockIdx.x * NWAVE + wave) * MTILE;
  const long long* startIdx = eidx;
  const long long* endIdx   = eidx + N_EDGES_C;
  const float bout = b_out[0];

  // ---- Phase 1: gather concat(x[start], x[end]) rows -> bf16 A-tile ----
  {
    const int c4 = row * 4;
#pragma unroll 4
    for (int r = 0; r < MTILE; ++r) {
      long long e = tileBase + r; if (e >= N_EDGES_C) e = N_EDGES_C - 1;
      long long node = hi ? endIdx[e] : startIdx[e];
      const float4 v = *(const float4*)(x + node * 64 + c4);
      v4bf b = { (__bf16)v.x, (__bf16)v.y, (__bf16)v.z, (__bf16)v.w };
      *(v4bf*)(hb + r * BSTRIDE + hi * 64 + c4) = b;
    }
  }

  for (int l = 0; l < NLAY; ++l) {
    // ---- block-load this layer's bf16 W^T into LDS with padded rows ----
    __syncthreads();  // all waves done reading sWt of previous layer
    {
      const uint4* src = (const uint4*)(WtRaw + (size_t)l * HID_C * HID_C);
      char* dstBase = (char*)sWt;
      for (int i = tid; i < (HID_C * HID_C) / 8; i += NWAVE * 32) {
        const int rw = i >> 4;          // W row (output column n)
        const int c  = i & 15;          // 16-byte chunk within row
        *(uint4*)(dstBase + rw * (WSTRIDE * 2) + c * 16) = src[i];
      }
    }
    __syncthreads();

    // ---- A fragments (contiguous k order, same as B: lanes<16 K{0-15}, lanes>=16 K{16-31}) ----
    v16bf A[2][4];
#pragma unroll
    for (int m = 0; m < 2; ++m) {
      const __bf16* hr = hb + (m * 16 + row) * BSTRIDE;
#pragma unroll
      for (int kc = 0; kc < 4; ++kc) {
        const int kb = kc * 32;
        v8bf lo8 = *(const v8bf*)(hr + kb + hi * 16);
        v8bf hi8 = *(const v8bf*)(hr + kb + hi * 16 + 8);
        A[m][kc] = cat16(lo8, hi8);
      }
    }

    // ---- hoist per-lane biases out of the GEMM loop ----
    float biasv[8];
#pragma unroll
    for (int nt = 0; nt < 8; ++nt) biasv[nt] = bs[l * HID_C + nt * 16 + row];

    // ---- GEMM with software-pipelined B-fragment loads ----
    // B 32x16 bf16: lane = column N, lanes<16 hold K 0-15, lanes>=16 hold K 16-31
    v8bf cLo = *(const v8bf*)(sWt + row * WSTRIDE + hi * 16);
    v8bf cHi = *(const v8bf*)(sWt + row * WSTRIDE + hi * 16 + 8);
    v8f acc0, acc1;
#pragma unroll
    for (int it = 0; it < 32; ++it) {
      const int nt = it >> 2, kc = it & 3;
      v8bf nLo = cLo, nHi = cHi;
      if (it < 31) {  // prefetch next (nt,kc) B fragment before issuing WMMAs
        const int nt2 = (it + 1) >> 2, kc2 = (it + 1) & 3;
        const __bf16* wp = sWt + (nt2 * 16 + row) * WSTRIDE + kc2 * 32 + hi * 16;
        nLo = *(const v8bf*)wp;
        nHi = *(const v8bf*)(wp + 8);
      }
      v16bf B = cat16(cLo, cHi);
      if (kc == 0) {
        acc0 = (v8f){0.f,0.f,0.f,0.f,0.f,0.f,0.f,0.f};
        acc1 = (v8f){0.f,0.f,0.f,0.f,0.f,0.f,0.f,0.f};
      }
      acc0 = __builtin_amdgcn_wmma_f32_16x16x32_bf16(false, A[0][kc], false, B,
                                                     (short)0, acc0, false, false);
      acc1 = __builtin_amdgcn_wmma_f32_16x16x32_bf16(false, A[1][kc], false, B,
                                                     (short)0, acc1, false, false);
      if (kc == 3) {
        // C layout: VGPR r -> M = r + 8*hi, N = lane&15 ; scatter + bias into f32 tile
        const int col = nt * 16 + row;
        const float bias = biasv[nt];
#pragma unroll
        for (int r = 0; r < 8; ++r) {
          hf[(r + hi * 8) * FSTRIDE + col]      = acc0[r] + bias;
          hf[(16 + r + hi * 8) * FSTRIDE + col] = acc1[r] + bias;
        }
      }
      cLo = nLo; cHi = nHi;
    }

    // ---- LayerNorm + tanh per row (wave-cooperative, float4 per lane) ----
    const float4 g4  = ((const float4*)(gammas + l * HID_C))[lane];
    const float4 be4 = ((const float4*)(betas + l * HID_C))[lane];
    const float4 w4  = ((const float4*)w_out)[lane];
    const bool last = (l == NLAY - 1);
#pragma unroll 2
    for (int r = 0; r < MTILE; ++r) {
      float4 v = ((const float4*)(hf + r * FSTRIDE))[lane];
      float s  = v.x + v.y + v.z + v.w;
      float sq = v.x * v.x + v.y * v.y + v.z * v.z + v.w * v.w;
      s  = wave_sum32(s);
      sq = wave_sum32(sq);
      const float mean = s * (1.0f / HID_C);
      const float var  = sq * (1.0f / HID_C) - mean * mean;
      const float rstd = rsqrtf(var + LNEPS);
      const float y0 = tanhf((v.x - mean) * rstd * g4.x + be4.x);
      const float y1 = tanhf((v.y - mean) * rstd * g4.y + be4.y);
      const float y2 = tanhf((v.z - mean) * rstd * g4.z + be4.z);
      const float y3 = tanhf((v.w - mean) * rstd * g4.w + be4.w);
      if (last) {
        float d = wave_sum32(y0 * w4.x + y1 * w4.y + y2 * w4.z + y3 * w4.w);
        const long long e = tileBase + r;
        if (lane == 0 && e < N_EDGES_C) out[e] = d + bout;
      } else {
        v4bf y = { (__bf16)y0, (__bf16)y1, (__bf16)y2, (__bf16)y3 };
        *(v4bf*)(hb + r * BSTRIDE + lane * 4) = y;   // next layer's A source
      }
    }
  }
}

extern "C" void kernel_launch(void* const* d_in, const int* in_sizes, int n_in,
                              void* d_out, int out_size, void* d_ws, size_t ws_size,
                              hipStream_t stream) {
  const float*     x      = (const float*)d_in[0];
  const long long* eidx   = (const long long*)d_in[1];
  const float*     Ws     = (const float*)d_in[2];
  const float*     bs     = (const float*)d_in[3];
  const float*     gammas = (const float*)d_in[4];
  const float*     betas  = (const float*)d_in[5];
  const float*     w_out  = (const float*)d_in[6];
  const float*     b_out  = (const float*)d_in[7];
  float*           out    = (float*)d_out;
  unsigned short*  wt     = (unsigned short*)d_ws;   // [L][N][K] bf16 bits (96 KB)

  // prep: W -> bf16 transposed
  {
    const int n = NLAY * HID_C * HID_C;
    prep_w_bf16<<<(n + 255) / 256, 256, 0, stream>>>(Ws, wt);
  }

  // main: 4 waves/block, 32 edges/wave -> 128 edges/block
  const long long tiles  = (N_EDGES_C + MTILE - 1) / MTILE;     // 25000
  const int       blocks = (int)((tiles + NWAVE - 1) / NWAVE);  // 6250
  const size_t    smem   = (size_t)HID_C * WSTRIDE * 2          // padded W^T
                         + (size_t)NWAVE * MTILE * FSTRIDE * 4  // f32 h tiles
                         + (size_t)NWAVE * MTILE * BSTRIDE * 2; // bf16 h tiles
  (void)hipFuncSetAttribute((const void*)edge_net_kernel,
                            hipFuncAttributeMaxDynamicSharedMemorySize, (int)smem);
  edge_net_kernel<<<blocks, dim3(NWAVE * 32), smem, stream>>>(
      x, eidx, wt, bs, gammas, betas, w_out, b_out, out);
}
